// AttentionHead_22084721836872
// MI455X (gfx1250) — compile-verified, matrix operand order checked by reading, not run on hardware
//
#include <hip/hip_runtime.h>

#define BB 8
#define SS 1024
#define HH 1024
#define DD 64

typedef _Float16 v16h __attribute__((ext_vector_type(16)));
typedef _Float16 v8h  __attribute__((ext_vector_type(8)));
typedef float    v8f  __attribute__((ext_vector_type(8)));
typedef float    v4f  __attribute__((ext_vector_type(4)));

union V16HU { v16h v; v8h h[2]; };
union V8FU  { v8f v; v4f q[2]; float f[8]; };

// Convert 8 consecutive f32 to packed f16 (vector loads underneath).
__device__ __forceinline__ v8h cvt8_f32(const float* p) {
  v4f a = *(const v4f*)p;
  v4f b = *(const v4f*)(p + 4);
  v8h r;
#pragma unroll
  for (int i = 0; i < 4; ++i) { r[i] = (_Float16)a[i]; r[i + 4] = (_Float16)b[i]; }
  return r;
}

// WMMA f16 16x16x32 operand loader from a row-major [16 x K] tile.
// A operand: lane L holds row (L&15); lanes 0-15 K = c0+{0..7,16..23},
// lanes 16-31 K = c0+{8..15,24..31}.  B operand loads identically from B^T.
__device__ __forceinline__ v16h load_op_f16(const _Float16* row, int c0, int lane) {
  int base = c0 + ((lane & 16) ? 8 : 0);
  V16HU u;
  u.h[0] = *(const v8h*)(row + base);
  u.h[1] = *(const v8h*)(row + base + 16);
  return u.v;
}
__device__ __forceinline__ v16h load_op_f32(const float* row, int c0, int lane) {
  int base = c0 + ((lane & 16) ? 8 : 0);
  V16HU u;
  u.h[0] = cvt8_f32(row + base);
  u.h[1] = cvt8_f32(row + base + 16);
  return u.v;
}

#define WMMA(a, b, c) \
  __builtin_amdgcn_wmma_f32_16x16x32_f16(false, (a), false, (b), (short)0, (c), false, false)

// ---------------------------------------------------------------------------
// Kernel 1: projections  q/k = X @ W^T + bias (f16, row-major [B*S][64]),
//           v stored transposed vT[b][d][s] for later WMMA A-operands.
// Tile computed transposed (M = d, N = rows) so f16 stores are contiguous.
// Dual accumulator chains hide the f16-WMMA->VALU WAR hazard slots.
// ---------------------------------------------------------------------------
__global__ void __launch_bounds__(256) proj_kernel(
    const float* __restrict__ q_in, const float* __restrict__ k_in,
    const float* __restrict__ v_in,
    const float* __restrict__ Wq, const float* __restrict__ bq,
    const float* __restrict__ Wk, const float* __restrict__ bk,
    const float* __restrict__ Wv, const float* __restrict__ bv,
    _Float16* __restrict__ qf, _Float16* __restrict__ kf,
    _Float16* __restrict__ vT)
{
  const int lane = threadIdx.x & 31;
  const int wid  = blockIdx.x * 8 + (threadIdx.x >> 5);   // 6144 waves
  const int ntile = wid & 3;                              // 4 d-tiles
  const int rtile = (wid >> 2) & 511;                     // 512 row tiles
  const int p     = wid >> 11;                            // 0=q 1=k 2=v

  const float* X    = (p == 0) ? q_in : (p == 1) ? k_in : v_in;
  const float* W    = (p == 0) ? Wq   : (p == 1) ? Wk   : Wv;
  const float* bias = (p == 0) ? bq   : (p == 1) ? bk   : bv;

  const int n0 = ntile * 16, r0 = rtile * 16;
  const float* wrow = W + (size_t)(n0 + (lane & 15)) * HH;  // A: M = d
  const float* xrow = X + (size_t)(r0 + (lane & 15)) * HH;  // B^T: N = row

  v8f acc0 = {}, acc1 = {};                // two independent WMMA chains
#pragma unroll 2
  for (int ks = 0; ks < HH; ks += 64) {
    acc0 = WMMA(load_op_f32(wrow, ks,      lane),
                load_op_f32(xrow, ks,      lane), acc0);
    acc1 = WMMA(load_op_f32(wrow, ks + 32, lane),
                load_op_f32(xrow, ks + 32, lane), acc1);
  }
  v8f acc = acc0 + acc1;

  const int nb = n0 + ((lane & 16) ? 8 : 0);   // this lane's 8 consecutive d
  V8FU af; af.v = acc;
#pragma unroll
  for (int i = 0; i < 8; ++i) af.f[i] += bias[nb + i];

  const int r = r0 + (lane & 15);
  v8h o;
#pragma unroll
  for (int i = 0; i < 8; ++i) o[i] = (_Float16)af.f[i];

  if (p < 2) {
    _Float16* dst = (p == 0) ? qf : kf;
    *(v8h*)(dst + (size_t)r * DD + nb) = o;    // 16B contiguous f16 store
  } else {
    const int b = r >> 10, s = r & 1023;       // vT[b][d][s]
#pragma unroll
    for (int i = 0; i < 8; ++i)
      vT[((size_t)(b * DD + nb + i)) * SS + s] = o[i];
  }
}

// ---------------------------------------------------------------------------
// Kernel 2: attn_1 -> scores[b][q][k] (fp32).  Computed as transposed tile
// (A = k rows, B = q rows) so each lane stores 8 consecutive k floats.
// ---------------------------------------------------------------------------
__global__ void __launch_bounds__(256) attn1_kernel(
    const _Float16* __restrict__ qf, const _Float16* __restrict__ kf,
    float* __restrict__ scores)
{
  const int lane = threadIdx.x & 31;
  const int wid  = blockIdx.x * 8 + (threadIdx.x >> 5);   // 2048 waves
  const int kc = wid & 3;                                 // 4 chunks of 256 k
  const int qt = (wid >> 2) & 63;
  const int b  = wid >> 8;
  const int q0 = qt * 16;

  const _Float16* qrow = qf + ((size_t)(b * SS + q0 + (lane & 15))) * DD;
  v16h B0 = load_op_f16(qrow, 0, lane);
  v16h B1 = load_op_f16(qrow, 32, lane);

  float* srow = scores + ((size_t)(b * SS + q0 + (lane & 15))) * SS
              + ((lane & 16) ? 8 : 0);
#pragma unroll 2
  for (int kt = 0; kt < 16; ++kt) {
    const int k0 = kc * 256 + kt * 16;
    const _Float16* krow = kf + ((size_t)(b * SS + k0 + (lane & 15))) * DD;
    v8f acc = {};
    acc = WMMA(load_op_f16(krow, 0, lane),  B0, acc);
    acc = WMMA(load_op_f16(krow, 32, lane), B1, acc);
    V8FU u; u.v = acc;
    *(v4f*)(srow + k0)     = u.q[0];
    *(v4f*)(srow + k0 + 4) = u.q[1];
  }
}

// ---------------------------------------------------------------------------
// Kernel 3: attn_2 accumulate.  Per query q: [16k x 64d] k_bias tile (A, fp32
// streamed + prefetch) x q-vectors of all 8 batches (B, N=8 padded to 16).
// RMW into scores with contiguous float4s (lanes with batch>=8 discarded).
// Unrolled x2: independent per-tile accumulators interleave WMMA chains.
// ---------------------------------------------------------------------------
__global__ void __launch_bounds__(256) attn2_kernel(
    const _Float16* __restrict__ qf, const float* __restrict__ kbias,
    float* __restrict__ scores)
{
  const int lane = threadIdx.x & 31;
  const int wid  = blockIdx.x * 8 + (threadIdx.x >> 5);   // 4096 waves
  const int kc = wid & 3;
  const int q  = wid >> 2;

  int bb = lane & 15; if (bb > 7) bb = 7;                  // clamp pad lanes
  const _Float16* qrow = qf + ((size_t)(bb * SS + q)) * DD;
  v16h B0 = load_op_f16(qrow, 0, lane);
  v16h B1 = load_op_f16(qrow, 32, lane);

  const int bvalid = ((lane & 15) < 8);
  float* srow = scores + ((size_t)((lane & 15) * SS + q)) * SS
              + ((lane & 16) ? 8 : 0);

#pragma unroll 2
  for (int kt = 0; kt < 16; ++kt) {
    const int k0 = kc * 256 + kt * 16;
    const float* arow = kbias + ((size_t)q * SS + k0 + (lane & 15)) * DD;
    __builtin_prefetch(arow + 16 * DD, 0, 0);              // next k-tile row
    v8f acc = {};
    acc = WMMA(load_op_f32(arow, 0, lane),  B0, acc);
    acc = WMMA(load_op_f32(arow, 32, lane), B1, acc);
    if (bvalid) {
      V8FU u; u.v = acc;
      v4f s0 = *(const v4f*)(srow + k0);
      v4f s1 = *(const v4f*)(srow + k0 + 4);
      s0 += u.q[0]; s1 += u.q[1];
      *(v4f*)(srow + k0)     = s0;
      *(v4f*)(srow + k0 + 4) = s1;
    }
  }
}

// ---------------------------------------------------------------------------
// Kernel 4: scale + mask + softmax (fp32), write f16 probabilities.
// One 256-thread block per (b,q) row of 1024.
// ---------------------------------------------------------------------------
__global__ void __launch_bounds__(256) softmax_kernel(
    const float* __restrict__ scores, const int* __restrict__ mask,
    _Float16* __restrict__ wf)
{
  __shared__ float red[8];
  const int tid = threadIdx.x;
  const size_t base = (size_t)blockIdx.x * SS;

  float v[4];
#pragma unroll
  for (int t = 0; t < 4; ++t) {
    const int idx = tid + t * 256;
    const float s = scores[base + idx];
    const int   m = mask[base + idx];
    v[t] = (m == 0) ? -1e9f : s * 0.125f;   // /sqrt(64), then mask
  }
  float mx = fmaxf(fmaxf(v[0], v[1]), fmaxf(v[2], v[3]));
#pragma unroll
  for (int off = 16; off; off >>= 1) mx = fmaxf(mx, __shfl_xor(mx, off, 32));
  if ((tid & 31) == 0) red[tid >> 5] = mx;
  __syncthreads();
  mx = red[0];
#pragma unroll
  for (int i = 1; i < 8; ++i) mx = fmaxf(mx, red[i]);
  __syncthreads();

  float p[4], sum = 0.f;
#pragma unroll
  for (int t = 0; t < 4; ++t) { p[t] = __expf(v[t] - mx); sum += p[t]; }
#pragma unroll
  for (int off = 16; off; off >>= 1) sum += __shfl_xor(sum, off, 32);
  if ((tid & 31) == 0) red[tid >> 5] = sum;
  __syncthreads();
  sum = 0.f;
#pragma unroll
  for (int i = 0; i < 8; ++i) sum += red[i];
  const float inv = 1.0f / sum;
#pragma unroll
  for (int t = 0; t < 4; ++t)
    wf[base + tid + t * 256] = (_Float16)(p[t] * inv);
}

// ---------------------------------------------------------------------------
// Kernel 5: values_1 -> out (initializes out).  Transposed tile trick again:
// A = vT rows (M=d), B = w rows (N=q) -> contiguous float4 stores of out.
// 4 independent accumulator chains per K-step.
// ---------------------------------------------------------------------------
__global__ void __launch_bounds__(256) values1_kernel(
    const _Float16* __restrict__ wf, const _Float16* __restrict__ vT,
    float* __restrict__ out)
{
  const int lane = threadIdx.x & 31;
  const int wid  = blockIdx.x * 8 + (threadIdx.x >> 5);   // 512 waves
  const int qt = wid & 63;
  const int b  = wid >> 6;
  const int q0 = qt * 16;

  const _Float16* wrow  = wf + ((size_t)(b * SS + q0 + (lane & 15))) * SS;
  const _Float16* vrow0 = vT + ((size_t)(b * DD + (lane & 15))) * SS;

  v8f acc[4] = {{}, {}, {}, {}};
  for (int ks = 0; ks < SS; ks += 32) {
    v16h Bw = load_op_f16(wrow, ks, lane);
#pragma unroll
    for (int t = 0; t < 4; ++t) {
      v16h A = load_op_f16(vrow0 + (size_t)t * 16 * SS, ks, lane);
      acc[t] = WMMA(A, Bw, acc[t]);
    }
  }
  float* orow = out + ((size_t)(b * SS + q0 + (lane & 15))) * DD
              + ((lane & 16) ? 8 : 0);
#pragma unroll
  for (int t = 0; t < 4; ++t) {
    V8FU u; u.v = acc[t];
    *(v4f*)(orow + t * 16)     = u.q[0];
    *(v4f*)(orow + t * 16 + 4) = u.q[1];
  }
}

// ---------------------------------------------------------------------------
// Kernel 6: values_2, atomically added into out.  Per q: A = w[.,q,.] rows
// (M=batch, padded), B = v_bias[q] tile gathered per-lane (each e-step is a
// 64B-contiguous wave access; lines reused across d-tiles via cache).
// ---------------------------------------------------------------------------
__global__ void __launch_bounds__(256) values2_kernel(
    const _Float16* __restrict__ wf, const float* __restrict__ vbias,
    float* __restrict__ out)
{
  const int lane = threadIdx.x & 31;
  const int wid  = blockIdx.x * 8 + (threadIdx.x >> 5);   // 4096 waves
  const int kc = wid & 3;
  const int q  = wid >> 2;

  int bb = lane & 15; if (bb > 7) bb = 7;
  const _Float16* wrow = wf + ((size_t)(bb * SS + q)) * SS;

  v8f acc[4] = {{}, {}, {}, {}};
  for (int ks = 0; ks < 8; ++ks) {
    const int c0  = kc * 256 + ks * 32;
    v16h A = load_op_f16(wrow, c0, lane);
    const int kkb = c0 + ((lane & 16) ? 8 : 0);
    const float* bbase = vbias + ((size_t)q * SS + kkb) * DD + (lane & 15);
    __builtin_prefetch(bbase + (size_t)32 * DD, 0, 0);     // next k-step
#pragma unroll
    for (int t = 0; t < 4; ++t) {
      const float* bp = bbase + t * 16;
      V16HU u;
#pragma unroll
      for (int e = 0; e < 8; ++e) {
        u.h[0][e] = (_Float16)bp[(size_t)e * DD];
        u.h[1][e] = (_Float16)bp[(size_t)(e + 16) * DD];
      }
      acc[t] = WMMA(A, u.v, acc[t]);
    }
  }
  if (lane < 16) {                 // lanes 0-15 hold batches 0..7 (vgpr i)
#pragma unroll
    for (int t = 0; t < 4; ++t) {
      V8FU u; u.v = acc[t];
#pragma unroll
      for (int i = 0; i < 8; ++i) {
        float* p = out + ((size_t)(i * SS + q)) * DD + t * 16 + lane;
        unsafeAtomicAdd(p, u.f[i]);   // global_atomic_add_f32, no return
      }
    }
  }
}

// ---------------------------------------------------------------------------
extern "C" void kernel_launch(void* const* d_in, const int* in_sizes, int n_in,
                              void* d_out, int out_size, void* d_ws, size_t ws_size,
                              hipStream_t stream) {
  const float* queryIn = (const float*)d_in[0];
  const float* keyIn   = (const float*)d_in[1];
  const float* valueIn = (const float*)d_in[2];
  const int*   maskIn  = (const int*)  d_in[3];
  const float* kbias   = (const float*)d_in[4];
  const float* vbias   = (const float*)d_in[5];
  const float* Wq = (const float*)d_in[6];
  const float* bq = (const float*)d_in[7];
  const float* Wk = (const float*)d_in[8];
  const float* bk = (const float*)d_in[9];
  const float* Wv = (const float*)d_in[10];
  const float* bv = (const float*)d_in[11];
  float* out = (float*)d_out;

  // Workspace layout (51 MB total):
  //   [0,1MB)   qf  f16 [B*S][64]
  //   [1,2MB)   kf  f16 [B*S][64]
  //   [2,3MB)   vT  f16 [B][64][S]
  //   [3,19MB)  wf  f16 [B][S][S]
  //   [19,51MB) scores f32 [B][S][S]
  char* ws = (char*)d_ws;
  _Float16* qf = (_Float16*)(ws);
  _Float16* kf = (_Float16*)(ws + (size_t)(1u << 20));
  _Float16* vT = (_Float16*)(ws + (size_t)(2u << 20));
  _Float16* wf = (_Float16*)(ws + (size_t)(3u << 20));
  float* scores = (float*)(ws + (size_t)(19u << 20));

  proj_kernel   <<<768, 256, 0, stream>>>(queryIn, keyIn, valueIn,
                                          Wq, bq, Wk, bk, Wv, bv, qf, kf, vT);
  attn1_kernel  <<<256, 256, 0, stream>>>(qf, kf, scores);
  attn2_kernel  <<<512, 256, 0, stream>>>(qf, kbias, scores);
  softmax_kernel<<<8192, 256, 0, stream>>>(scores, maskIn, wf);
  values1_kernel<<<64,  256, 0, stream>>>(wf, vT, out);
  values2_kernel<<<512, 256, 0, stream>>>(wf, vbias, out);
}
